// LAttention_58737972740609
// MI455X (gfx1250) — compile-verified, matrix operand order checked by reading, not run on hardware
//
#include <hip/hip_runtime.h>
#include <hip/hip_bf16.h>

// ---------------------------------------------------------------------------
// LAttention for MI455X (gfx1250, wave32, WMMA, async global->LDS)
//   DIM=512, HEADS=8, D=64, B=2, S=128, N=256  ->  M = B*S*N = 65536 rows
//   Stage 0: convert x, Wq, Wk, Wv to bf16 (bf16 x = 64 MB -> L2-resident)
//   Stage 1: q,k,v = x @ W{q,k,v}^T  (bf16 WMMA, f32 acc, double-buffered
//            global_load_async_to_lds_b128 pipeline, s_wait_asynccnt fences)
//   Stage 2: per (b,s,h): softmax reductions + (kvi @ Wr^T + q) via WMMA
// ---------------------------------------------------------------------------

#define BDIM   2
#define SEQ    128
#define NTOK   256
#define DIMC   512
#define NHEADS 8
#define DH     64
#define MROWS  (BDIM * SEQ * NTOK)   // 65536
#define SCALE_F 0.04419417382415922f // 512^-0.5

typedef __attribute__((ext_vector_type(8)))  float   v8f;
typedef __attribute__((ext_vector_type(8)))  __bf16  v8bf;
typedef __attribute__((ext_vector_type(16))) __bf16  v16bf;

union Frag {
    v8bf  h[2];
    v16bf v;
};
union Pack8 {
    v8bf   v;
    __bf16 e[8];
};

__device__ __forceinline__ v8f wmma_bf16(v16bf a, v16bf b, v8f c) {
    return __builtin_amdgcn_wmma_f32_16x16x32_bf16(
        /*neg_a=*/false, a, /*neg_b=*/false, b,
        /*c_mod=*/(short)0, c, /*reuse_a=*/false, /*reuse_b=*/false);
}

// gfx1250: generic pointers into LDS carry the LDS byte offset in the low
// 32 bits (aperture in the high half), so truncation yields the DS address.
__device__ __forceinline__ uint32_t lds_addr(const void* p) {
    return (uint32_t)(uintptr_t)p;
}

// Async DMA of 16 bytes/lane from global memory into LDS (ASYNCcnt-tracked).
__device__ __forceinline__ void async_load_b128(uint32_t lds_off, const void* gptr) {
    const uint64_t ga = (uint64_t)(uintptr_t)gptr;
    asm volatile("global_load_async_to_lds_b128 %0, %1, off"
                 :: "v"(lds_off), "v"(ga)
                 : "memory");
}

// ---------------------------------------------------------------------------
// Stage 0: f32 -> bf16 conversion (8 elements / thread, b128 in / b128 out)
// ---------------------------------------------------------------------------
__global__ __launch_bounds__(256) void f32_to_bf16_kernel(
    const float* __restrict__ src, __bf16* __restrict__ dst, int n8) {
    const int i = blockIdx.x * 256 + threadIdx.x;
    if (i < n8) {
        const float4 f0 = ((const float4*)src)[(size_t)i * 2 + 0];
        const float4 f1 = ((const float4*)src)[(size_t)i * 2 + 1];
        Pack8 pk;
        pk.e[0] = (__bf16)f0.x; pk.e[1] = (__bf16)f0.y;
        pk.e[2] = (__bf16)f0.z; pk.e[3] = (__bf16)f0.w;
        pk.e[4] = (__bf16)f1.x; pk.e[5] = (__bf16)f1.y;
        pk.e[6] = (__bf16)f1.z; pk.e[7] = (__bf16)f1.w;
        *(v8bf*)(dst + (size_t)i * 8) = pk.v;
    }
}

// ---------------------------------------------------------------------------
// Stage 1: fused QKV projection GEMM (bf16 in, f32 out).
//   grid.x : M / 128 row tiles
//   grid.y : 24 = 3 weights * (512/64) column tiles
//   block  : 256 threads = 8 waves; wave w owns rows [w*16, w*16+16)
//   Double-buffered async-LDS pipeline: 3 async b128 ops / wave / K-tile.
// ---------------------------------------------------------------------------
__global__ __launch_bounds__(256) void qkv_gemm_kernel(
    const __bf16* __restrict__ Xb, const __bf16* __restrict__ Wb,
    float* __restrict__ Qb, float* __restrict__ Kb, float* __restrict__ Vb) {

    __shared__ __bf16 As[2][128 * 32];   // x tile [row][k], 8 KB per buffer
    __shared__ __bf16 Bs[2][64 * 32];    // W tile [col][k], 4 KB per buffer

    const int tid  = threadIdx.x;
    const int wave = tid >> 5;
    const int lane = tid & 31;
    const int g    = lane >> 4;          // K-group for frags, M-half for C
    const int l15  = lane & 15;

    const int mBase   = blockIdx.x * 128;
    const int colTile = blockIdx.y;              // 0..23
    const int wsel    = colTile >> 3;            // 0:q 1:k 2:v
    const int jBase   = (colTile & 7) * 64;

    float* O = (wsel == 0) ? Qb : (wsel == 1) ? Kb : Vb;
    const __bf16* Wrow = Wb + (size_t)(wsel * DIMC + jBase) * DIMC;

    // A tile: 128 rows x 64 B -> 512 x 16B chunks, 2 per thread.
    // B tile:  64 rows x 64 B -> 256 x 16B chunks, 1 per thread.
    auto issue_tile = [&](int t, int p) {
        const int k0 = t * 32;
        {
            const int c0 = tid, r0 = c0 >> 2, q0 = c0 & 3;
            async_load_b128(lds_addr(&As[p][r0 * 32 + q0 * 8]),
                            Xb + (size_t)(mBase + r0) * DIMC + k0 + q0 * 8);
            const int c1 = tid + 256, r1 = c1 >> 2, q1 = c1 & 3;
            async_load_b128(lds_addr(&As[p][r1 * 32 + q1 * 8]),
                            Xb + (size_t)(mBase + r1) * DIMC + k0 + q1 * 8);
        }
        {
            const int j = tid >> 2, q2 = tid & 3;
            async_load_b128(lds_addr(&Bs[p][j * 32 + q2 * 8]),
                            Wrow + (size_t)j * DIMC + k0 + q2 * 8);
        }
    };

    const v8f vzero = {0.f, 0.f, 0.f, 0.f, 0.f, 0.f, 0.f, 0.f};
    v8f acc[4];
#pragma unroll
    for (int nt = 0; nt < 4; ++nt) acc[nt] = vzero;

    issue_tile(0, 0);

    for (int t = 0; t < 16; ++t) {
        const int p = t & 1;
        if (t < 15) {
            issue_tile(t + 1, p ^ 1);
            // 6 outstanding; <=3 means the current tile's 3 copies landed
            asm volatile("s_wait_asynccnt 0x3" ::: "memory");
        } else {
            asm volatile("s_wait_asynccnt 0x0" ::: "memory");
        }
        __syncthreads();

        // bf16 A layout: element i of v16bf holds K = (i>>3)*16 + g*8 + (i&7)
        Frag a;
        a.h[0] = *(const v8bf*)&As[p][(wave * 16 + l15) * 32 + g * 8];
        a.h[1] = *(const v8bf*)&As[p][(wave * 16 + l15) * 32 + 16 + g * 8];
#pragma unroll
        for (int nt = 0; nt < 4; ++nt) {
            Frag b;
            b.h[0] = *(const v8bf*)&Bs[p][(nt * 16 + l15) * 32 + g * 8];
            b.h[1] = *(const v8bf*)&Bs[p][(nt * 16 + l15) * 32 + 16 + g * 8];
            acc[nt] = wmma_bf16(a.v, b.v, acc[nt]);
        }
        __syncthreads();   // all reads of buffer p done before it is refilled
    }

    // C layout: lane<16 -> M=r, lane>=16 -> M=8+r; N = lane&15
#pragma unroll
    for (int nt = 0; nt < 4; ++nt) {
#pragma unroll
        for (int r = 0; r < 8; ++r) {
            const int m   = mBase + wave * 16 + r + 8 * g;
            const int col = jBase + nt * 16 + l15;
            O[(size_t)m * DIMC + col] = acc[nt][r];
        }
    }
}

// ---------------------------------------------------------------------------
// Stage 2: per-(b,s,h) fused softmax reductions + Wr mix + residual.
//   grid.x : B*S*HEADS = 2048 blocks; block = 256 threads = 8 waves
// ---------------------------------------------------------------------------
__global__ __launch_bounds__(256) void attn_fuse_kernel(
    const float* __restrict__ Qb, const float* __restrict__ Kb,
    const float* __restrict__ Vb, const float* __restrict__ Wr,
    const float* __restrict__ wAlpha, const float* __restrict__ wBeta,
    float* __restrict__ out) {

    __shared__ float  gqS[DH];
    __shared__ float  gkS[DH];
    __shared__ __bf16 Akvi[NTOK * DH];  // 32 KB
    __shared__ __bf16 WrS[DH * DH];     //  8 KB

    const int tid  = threadIdx.x;
    const int wave = tid >> 5;
    const int lane = tid & 31;
    const int bsh  = blockIdx.x;
    const int h    = bsh & (NHEADS - 1);
    const int bs   = bsh >> 3;
    const size_t rowBase = (size_t)bs * NTOK;
    const int d0 = lane, d1 = lane + 32;

    if (tid < DH) { gqS[tid] = 0.f; gkS[tid] = 0.f; }

    // stage Wr (64x64 f32 -> bf16, row-major: row j contiguous in d = K)
    {
        const int j = tid >> 2;
        const int c = (tid & 3) * 16;
        const float* src = Wr + j * DH + c;
#pragma unroll
        for (int i = 0; i < 16; i += 4) {
            float4 f = *(const float4*)(src + i);
            __bf16* dst = &WrS[j * DH + c + i];
            dst[0] = (__bf16)f.x; dst[1] = (__bf16)f.y;
            dst[2] = (__bf16)f.z; dst[3] = (__bf16)f.w;
        }
    }
    const float wa0 = wAlpha[d0] * SCALE_F, wa1 = wAlpha[d1] * SCALE_F;
    const float wb0 = wBeta[d0] * SCALE_F, wb1 = wBeta[d1] * SCALE_F;
    __syncthreads();

    // ---- phase 1: global_query = sum_n q * softmax_d(q * w_alpha * scale)
    {
        float a0 = 0.f, a1 = 0.f;
        for (int n = wave; n < NTOK; n += 8) {
            const float* qr = Qb + (rowBase + n) * DIMC + h * DH;
            if (n + 8 < NTOK) __builtin_prefetch(qr + 8 * DIMC, 0, 1);
            float q0 = qr[d0], q1 = qr[d1];
            float t0 = q0 * wa0, t1 = q1 * wa1;
            float m = fmaxf(t0, t1);
#pragma unroll
            for (int off = 16; off > 0; off >>= 1)
                m = fmaxf(m, __shfl_xor(m, off, 32));
            float e0 = __expf(t0 - m), e1 = __expf(t1 - m);
            float s = e0 + e1;
#pragma unroll
            for (int off = 16; off > 0; off >>= 1)
                s += __shfl_xor(s, off, 32);
            float inv = 1.0f / s;
            a0 += q0 * e0 * inv;
            a1 += q1 * e1 * inv;
        }
        atomicAdd(&gqS[d0], a0);
        atomicAdd(&gqS[d1], a1);
    }
    __syncthreads();
    const float gq0 = gqS[d0], gq1 = gqS[d1];

    // ---- phase 2: p = gq*k ; global_key = sum_n p * softmax_d(p*w_beta*scale)
    {
        float a0 = 0.f, a1 = 0.f;
        for (int n = wave; n < NTOK; n += 8) {
            const float* kr = Kb + (rowBase + n) * DIMC + h * DH;
            float p0 = gq0 * kr[d0], p1 = gq1 * kr[d1];
            float t0 = p0 * wb0, t1 = p1 * wb1;
            float m = fmaxf(t0, t1);
#pragma unroll
            for (int off = 16; off > 0; off >>= 1)
                m = fmaxf(m, __shfl_xor(m, off, 32));
            float e0 = __expf(t0 - m), e1 = __expf(t1 - m);
            float s = e0 + e1;
#pragma unroll
            for (int off = 16; off > 0; off >>= 1)
                s += __shfl_xor(s, off, 32);
            float inv = 1.0f / s;
            a0 += p0 * e0 * inv;
            a1 += p1 * e1 * inv;
        }
        atomicAdd(&gkS[d0], a0);
        atomicAdd(&gkS[d1], a1);
    }
    __syncthreads();
    const float gk0 = gkS[d0], gk1 = gkS[d1];

    // ---- phase 3: kvi = gk * v -> bf16 in LDS (A matrix: [n][d], K=d) ----
    for (int n = wave; n < NTOK; n += 8) {
        const float* vr = Vb + (rowBase + n) * DIMC + h * DH;
        Akvi[n * DH + d0] = (__bf16)(gk0 * vr[d0]);
        Akvi[n * DH + d1] = (__bf16)(gk1 * vr[d1]);
    }
    __syncthreads();

    // ---- phase 4: result = kvi @ Wr^T + q via WMMA.
    //   M = 256 (n), K = 64 (d), Nout = 64. Wave w handles M-tiles w, w+8.
    const int g = lane >> 4, l15 = lane & 15;
    const v8f vzero = {0.f, 0.f, 0.f, 0.f, 0.f, 0.f, 0.f, 0.f};
    for (int mt = wave; mt < 16; mt += 8) {
        v8f acc[4];
#pragma unroll
        for (int nt = 0; nt < 4; ++nt) acc[nt] = vzero;
#pragma unroll
        for (int kb = 0; kb < 2; ++kb) {
            Frag a;
            a.h[0] = *(const v8bf*)&Akvi[(mt * 16 + l15) * DH + kb * 32 + g * 8];
            a.h[1] = *(const v8bf*)&Akvi[(mt * 16 + l15) * DH + kb * 32 + 16 + g * 8];
#pragma unroll
            for (int nt = 0; nt < 4; ++nt) {
                Frag b;
                b.h[0] = *(const v8bf*)&WrS[(nt * 16 + l15) * DH + kb * 32 + g * 8];
                b.h[1] = *(const v8bf*)&WrS[(nt * 16 + l15) * DH + kb * 32 + 16 + g * 8];
                acc[nt] = wmma_bf16(a.v, b.v, acc[nt]);
            }
        }
        // epilogue: residual + output in 'b s n (h d)' layout
#pragma unroll
        for (int nt = 0; nt < 4; ++nt) {
#pragma unroll
            for (int r = 0; r < 8; ++r) {
                const int n = mt * 16 + r + 8 * g;
                const size_t idx =
                    (rowBase + n) * DIMC + h * DH + nt * 16 + l15;
                out[idx] = acc[nt][r] + Qb[idx];
            }
        }
    }
}

// ---------------------------------------------------------------------------
extern "C" void kernel_launch(void* const* d_in, const int* in_sizes, int n_in,
                              void* d_out, int out_size, void* d_ws, size_t ws_size,
                              hipStream_t stream) {
    (void)in_sizes; (void)n_in; (void)out_size; (void)ws_size;

    const float* x      = (const float*)d_in[0];
    const float* Wq     = (const float*)d_in[1];
    const float* Wk     = (const float*)d_in[2];
    const float* Wv     = (const float*)d_in[3];
    const float* Wr     = (const float*)d_in[4];
    const float* wAlpha = (const float*)d_in[5];
    const float* wBeta  = (const float*)d_in[6];
    float* out = (float*)d_out;

    // workspace layout:
    //   Xb  : bf16 [M, 512]          64 MB (L2-resident on MI455X)
    //   Wbb : bf16 [3*512, 512]      1.5 MB
    //   Q/K/V : f32 [M, 512] each    3 x 128 MB
    __bf16* Xb  = (__bf16*)d_ws;
    __bf16* Wbb = Xb + (size_t)MROWS * DIMC;
    float*  Qb  = (float*)(Wbb + (size_t)3 * DIMC * DIMC);
    float*  Kb  = Qb + (size_t)MROWS * DIMC;
    float*  Vb  = Kb + (size_t)MROWS * DIMC;

    // Stage 0: bf16 packing
    const int n8x = MROWS * DIMC / 8;           // 4194304
    const int n8w = DIMC * DIMC / 8;            // 32768
    f32_to_bf16_kernel<<<n8x / 256, 256, 0, stream>>>(x,  Xb, n8x);
    f32_to_bf16_kernel<<<n8w / 256, 256, 0, stream>>>(Wq, Wbb + (size_t)0 * DIMC * DIMC, n8w);
    f32_to_bf16_kernel<<<n8w / 256, 256, 0, stream>>>(Wk, Wbb + (size_t)1 * DIMC * DIMC, n8w);
    f32_to_bf16_kernel<<<n8w / 256, 256, 0, stream>>>(Wv, Wbb + (size_t)2 * DIMC * DIMC, n8w);

    // Stage 1: QKV GEMM
    dim3 gridA(MROWS / 128, 24);   // 24 = 3 weights * 8 column tiles
    qkv_gemm_kernel<<<gridA, 256, 0, stream>>>(Xb, Wbb, Qb, Kb, Vb);

    // Stage 2: fused attention reductions + mix + residual
    dim3 gridB(BDIM * SEQ * NHEADS);  // 2048
    attn_fuse_kernel<<<gridB, 256, 0, stream>>>(Qb, Kb, Vb, Wr, wAlpha, wBeta, out);
}
